// PointNet2Segmentation_34239479284301
// MI455X (gfx1250) — matrix operand order repaired and checked by reading, not compiled, not run on hardware
//
#include <hip/hip_runtime.h>
#include <hip/hip_bf16.h>

typedef _Float16 h8   __attribute__((ext_vector_type(8)));
typedef _Float16 v16h __attribute__((ext_vector_type(16)));
typedef float    v8f  __attribute__((ext_vector_type(8)));

static constexpr int kB  = 16;
static constexpr int kN  = 4096;
static constexpr int kM1 = 820;
static constexpr int kM2 = 205;
static constexpr int kK  = 64;   // max neighbors

// ---------------------------------------------------------------- descriptors
struct Layer {
  const _Float16* wpk;   // packed B-fragment weights
  const float*    bias;  // f32 bias (original param)
  int dinPad, dout, doutPad, relu;
};
struct MlpDesc { int nL; Layer L[6]; };

// ---------------------------------------------------------------- WMMA core
// Activations: LDS row-major f16, row stride = dinPad (multiple of 32).
// A frag (16x32 f16): lane<16 -> M=lane, K = 0..15 ; lane>=16 -> K=16..31.
// B frag packed in global as [ct][kt][lane][16 halves]: lane<16 K=kt*32+0..15 (N=lane),
//   lane>=16 K=kt*32+16..31 (N=lane-16), element e -> K = base+e.
// C frag (16x16 f32): VGPR v -> row = v + 8*(lane>=16), col = lane&15.
__device__ __forceinline__ void run_layer_rows(const _Float16* sin, _Float16* sout,
                                               const Layer L, int rtiles, int tid)
{
  const int lane = tid & 31;
  const int wave = tid >> 5;
  const int nw   = 4;
  const int ctiles = (L.dout + 15) >> 4;
  const int ktiles = L.dinPad >> 5;
  for (int t = wave; t < rtiles * ctiles; t += nw) {
    const int rt = t / ctiles;
    const int ct = t - rt * ctiles;
    v8f acc = {};
    const _Float16* arow = sin + (size_t)(rt * 16 + (lane & 15)) * L.dinPad;
    const int kb = (lane >> 4) << 4;
    const _Float16* wp = L.wpk + ((size_t)ct * ktiles * 32 + lane) * 16;
    for (int kt = 0; kt < ktiles; ++kt) {
      h8 lo = *(const h8*)(arow + kt * 32 + kb);
      h8 hi = *(const h8*)(arow + kt * 32 + kb + 8);
      v16h a = __builtin_shufflevector(lo, hi, 0,1,2,3,4,5,6,7,8,9,10,11,12,13,14,15);
      v16h b = *(const v16h*)(wp);
      wp += 32 * 16;
      acc = __builtin_amdgcn_wmma_f32_16x16x32_f16(false, a, false, b, (short)0, acc,
                                                   false, false);
    }
    const int col   = ct * 16 + (lane & 15);
    const float bv  = (col < L.dout) ? L.bias[col] : 0.0f;
    const int rbase = rt * 16 + ((lane >> 4) << 3);
#pragma unroll
    for (int v = 0; v < 8; ++v) {
      float xv = acc[v] + bv;
      if (L.relu) xv = fmaxf(xv, 0.0f);
      sout[(size_t)(rbase + v) * L.doutPad + col] = (_Float16)xv;
    }
  }
}

// ---------------------------------------------------------------- weight packing
__global__ __launch_bounds__(128) void pack_w_kernel(const float* __restrict__ W,
                                                     int din, int dout,
                                                     _Float16* __restrict__ dst)
{
  const int ktiles = (din + 31) / 32;
  const int ctiles = (dout + 15) / 16;
  const int total  = ctiles * ktiles * 32;
  const int u = blockIdx.x * 128 + threadIdx.x;
  if (u >= total) return;
  const int lane = u & 31;
  const int t    = u >> 5;
  const int kt   = t % ktiles;
  const int ct   = t / ktiles;
  const int kb   = kt * 32 + ((lane >> 4) << 4);
  const int n    = ct * 16 + (lane & 15);
  _Float16* d = dst + (size_t)u * 16;
#pragma unroll
  for (int e = 0; e < 16; ++e) {
    const int k = kb + e;
    float v = (k < din && n < dout) ? W[(size_t)k * dout + n] : 0.0f;
    d[e] = (_Float16)v;
  }
}

// ---------------------------------------------------------------- FPS (per-sample block)
__global__ __launch_bounds__(256) void fps_kernel(const float* __restrict__ pos,
                                                  int n, int M,
                                                  float* __restrict__ centers)
{
  const int b = blockIdx.x;
  const float* p = pos + (size_t)b * n * 3;
  float* cent    = centers + (size_t)b * M * 3;
  __shared__ float dmin[4096];
  __shared__ float rv[256];
  __shared__ int   ri[256];
  __shared__ int   s_last;
  const int tid = threadIdx.x;
  for (int i = tid; i < n; i += 256) dmin[i] = INFINITY;
  if (tid == 0) s_last = 0;
  __syncthreads();
  for (int m = 0; m < M; ++m) {
    const int last = s_last;
    const float lx = p[last * 3 + 0], ly = p[last * 3 + 1], lz = p[last * 3 + 2];
    if (tid == 0) { cent[m * 3 + 0] = lx; cent[m * 3 + 1] = ly; cent[m * 3 + 2] = lz; }
    float bestv = -INFINITY; int besti = 0x7fffffff;
    for (int i = tid; i < n; i += 256) {
      float dx = p[i * 3 + 0] - lx, dy = p[i * 3 + 1] - ly, dz = p[i * 3 + 2] - lz;
      float d2 = dx * dx + dy * dy + dz * dz;
      float dm = fminf(dmin[i], d2);
      dmin[i] = dm;
      if (dm > bestv) { bestv = dm; besti = i; }   // strided -> keeps lowest idx on tie
    }
    rv[tid] = bestv; ri[tid] = besti;
    __syncthreads();
    for (int s = 128; s > 0; s >>= 1) {
      if (tid < s) {
        float v2 = rv[tid + s]; int i2 = ri[tid + s];
        if (v2 > rv[tid] || (v2 == rv[tid] && i2 < ri[tid])) { rv[tid] = v2; ri[tid] = i2; }
      }
      __syncthreads();
    }
    if (tid == 0) s_last = ri[0];
    __syncthreads();
  }
}

// ---------------------------------------------------------------- ball query (wave32/center)
__global__ __launch_bounds__(128) void ball_kernel(const float* __restrict__ pts,
                                                   const float* __restrict__ cents,
                                                   int n, int M, float r2,
                                                   int* __restrict__ gidx,
                                                   int* __restrict__ nvArr)
{
  const int b    = blockIdx.y;
  const int m    = blockIdx.x * 4 + (threadIdx.x >> 5);
  const int lane = threadIdx.x & 31;
  if (m >= M) return;
  const float* p = pts + (size_t)b * n * 3;
  const float* c = cents + ((size_t)b * M + m) * 3;
  const float cx = c[0], cy = c[1], cz = c[2];
  int* g = gidx + ((size_t)b * M + m) * kK;
  int cnt = 0;
  for (int j0 = 0; j0 < n && cnt < kK; j0 += 32) {
    const int j = j0 + lane;
    bool in = false;
    if (j < n) {
      float dx = p[j * 3 + 0] - cx, dy = p[j * 3 + 1] - cy, dz = p[j * 3 + 2] - cz;
      in = (dx * dx + dy * dy + dz * dz) <= r2;
    }
    unsigned mask = (unsigned)__ballot(in);
    int slot = cnt + __popc(mask & ((1u << lane) - 1u));
    if (in && slot < kK) g[slot] = j;
    cnt += __popc(mask);
  }
  if (lane == 0) nvArr[(size_t)b * M + m] = (cnt < kK) ? cnt : kK;
}

// ---------------------------------------------------------------- fused PointConv
template <int CAPA, int CAPB>
__global__ __launch_bounds__(128) void pointconv_kernel(
    const float* __restrict__ feats, int F, const float* __restrict__ pts,
    const float* __restrict__ cents, const int* __restrict__ gidx,
    const int* __restrict__ nvArr, int n, int M, MlpDesc desc,
    float* __restrict__ outF)
{
  __shared__ _Float16 sA[64 * CAPA];
  __shared__ _Float16 sB[64 * CAPB];
  const int b = blockIdx.y, m = blockIdx.x, tid = threadIdx.x;
  const float* p  = pts   + (size_t)b * n * 3;
  const float* fb = feats + (size_t)b * n * F;
  const float* c  = cents + ((size_t)b * M + m) * 3;
  const int*   g  = gidx  + ((size_t)b * M + m) * kK;
  const int    nv = nvArr[(size_t)b * M + m];
  const int  din0 = desc.L[0].dinPad;
  const float cx = c[0], cy = c[1], cz = c[2];
  for (int e = tid; e < 64 * din0; e += 128) {
    const int k = e / din0, col = e - k * din0;
    float v = 0.0f;
    if (k < nv) {
      const int src = g[k];
      if (col < F) v = fb[(size_t)src * F + col];
      else if (col < F + 3) {
        const int d = col - F;
        v = p[(size_t)src * 3 + d] - (d == 0 ? cx : (d == 1 ? cy : cz));
      }
    }
    sA[e] = (_Float16)v;
  }
  __syncthreads();
  _Float16* bin  = sA;
  _Float16* bout = sB;
  for (int l = 0; l < desc.nL; ++l) {
    run_layer_rows(bin, bout, desc.L[l], 4, tid);
    __syncthreads();
    _Float16* t = bin; bin = bout; bout = t;
  }
  const Layer Ll = desc.L[desc.nL - 1];
  const _Float16* fo = bin;   // last layer's output
  for (int col = tid; col < Ll.dout; col += 128) {
    float mx = -INFINITY;
    for (int k = 0; k < nv; ++k)
      mx = fmaxf(mx, (float)fo[(size_t)k * Ll.doutPad + col]);
    outF[(((size_t)b * M) + m) * Ll.dout + col] = mx;
  }
}

// ---------------------------------------------------------------- row-strip MLP
template <int CAPA, int CAPB>
__global__ __launch_bounds__(128) void mlp_rows_kernel(
    const _Float16* __restrict__ gIn, size_t inSampleStride, int rowsTotal,
    MlpDesc desc, _Float16* __restrict__ gOutH, size_t outHSampleStride,
    float* __restrict__ gOutF, size_t outFSampleStride, int outFStride, int outFCols)
{
  __shared__ _Float16 sA[16 * CAPA];
  __shared__ _Float16 sB[16 * CAPB];
  const int b = blockIdx.y, strip = blockIdx.x, tid = threadIdx.x;
  const int r0 = strip * 16;
  int vrows = rowsTotal - r0; if (vrows > 16) vrows = 16;
  const int din0 = desc.L[0].dinPad;
  const _Float16* gi = gIn + (size_t)b * inSampleStride + (size_t)r0 * din0;
  for (int e = tid; e < 16 * din0; e += 128) {
    const int r = e / din0;
    sA[e] = (r < vrows) ? gi[e] : (_Float16)0.0f;
  }
  __syncthreads();
  _Float16* bin  = sA;
  _Float16* bout = sB;
  for (int l = 0; l < desc.nL; ++l) {
    run_layer_rows(bin, bout, desc.L[l], 1, tid);
    __syncthreads();
    _Float16* t = bin; bin = bout; bout = t;
  }
  const Layer Ll = desc.L[desc.nL - 1];
  const _Float16* fo = bin;   // last layer's output
  if (gOutH) {
    _Float16* go = gOutH + (size_t)b * outHSampleStride + (size_t)r0 * Ll.doutPad;
    for (int e = tid; e < vrows * Ll.doutPad; e += 128) go[e] = fo[e];
  }
  if (gOutF) {
    float* go = gOutF + (size_t)b * outFSampleStride + (size_t)r0 * outFStride;
    for (int e = tid; e < vrows * outFCols; e += 128) {
      const int r = e / outFCols, col = e - r * outFCols;
      go[(size_t)r * outFStride + col] = (float)fo[(size_t)r * Ll.doutPad + col];
    }
  }
}

// ---------------------------------------------------------------- builders / reductions
__global__ __launch_bounds__(256) void build_sa3_in(const float* __restrict__ x2,
                                                    const float* __restrict__ p2,
                                                    _Float16* __restrict__ out)
{
  const size_t total = (size_t)kB * kM2 * 288;
  size_t i = (size_t)blockIdx.x * 256 + threadIdx.x;
  if (i >= total) return;
  const int c = (int)(i % 288);
  const size_t rb = i / 288;
  const int r = (int)(rb % kM2), b = (int)(rb / kM2);
  float v = 0.0f;
  if (c < 256)      v = x2[((size_t)b * kM2 + r) * 256 + c];
  else if (c < 259) v = p2[((size_t)b * kM2 + r) * 3 + (c - 256)];
  out[i] = (_Float16)v;
}

__global__ __launch_bounds__(256) void build_fp3_in(const float* __restrict__ xg,
                                                    const float* __restrict__ x2,
                                                    _Float16* __restrict__ out)
{
  const size_t total = (size_t)kB * kM2 * 1280;
  size_t i = (size_t)blockIdx.x * 256 + threadIdx.x;
  if (i >= total) return;
  const int c = (int)(i % 1280);
  const size_t rb = i / 1280;
  const int r = (int)(rb % kM2), b = (int)(rb / kM2);
  float v = (c < 1024) ? xg[(size_t)b * 1024 + c]
                       : x2[((size_t)b * kM2 + r) * 256 + (c - 1024)];
  out[i] = (_Float16)v;
}

__global__ __launch_bounds__(256) void colmax_kernel(const _Float16* __restrict__ src,
                                                     int rows, int cols,
                                                     float* __restrict__ dst)
{
  const int b = blockIdx.x;
  const int c = blockIdx.y * 256 + threadIdx.x;
  if (c >= cols) return;
  const _Float16* s = src + (size_t)b * rows * cols + c;
  float mx = -INFINITY;
  for (int r = 0; r < rows; ++r) mx = fmaxf(mx, (float)s[(size_t)r * cols]);
  dst[(size_t)b * cols + c] = mx;
}

// kNN(3) inverse-d2 interpolation + skip concat -> f16 MLP input rows
__global__ __launch_bounds__(128) void interp_concat_kernel(
    const float* __restrict__ src, const float* __restrict__ psrc,
    const float* __restrict__ pdst, const float* __restrict__ skip,
    int ns, int nd, int Ds, int Dk, int dinPad, _Float16* __restrict__ out)
{
  const int b = blockIdx.y;
  const int i = blockIdx.x * 128 + threadIdx.x;
  if (i >= nd) return;
  const float* pd = pdst + ((size_t)b * nd + i) * 3;
  const float* ps = psrc + (size_t)b * ns * 3;
  const float dx0 = pd[0], dy0 = pd[1], dz0 = pd[2];
  float bd0 = INFINITY, bd1 = INFINITY, bd2 = INFINITY;
  int   bi0 = 0, bi1 = 0, bi2 = 0;
  for (int j = 0; j < ns; ++j) {
    const float dx = dx0 - ps[j * 3 + 0];
    const float dy = dy0 - ps[j * 3 + 1];
    const float dz = dz0 - ps[j * 3 + 2];
    const float d2 = dx * dx + dy * dy + dz * dz;
    if (d2 < bd0)      { bd2 = bd1; bi2 = bi1; bd1 = bd0; bi1 = bi0; bd0 = d2; bi0 = j; }
    else if (d2 < bd1) { bd2 = bd1; bi2 = bi1; bd1 = d2; bi1 = j; }
    else if (d2 < bd2) { bd2 = d2; bi2 = j; }
  }
  float w0 = 1.0f / fmaxf(bd0, 1e-16f);
  float w1 = 1.0f / fmaxf(bd1, 1e-16f);
  float w2 = 1.0f / fmaxf(bd2, 1e-16f);
  const float wsum = w0 + w1 + w2;
  w0 /= wsum; w1 /= wsum; w2 /= wsum;
  const float* s0 = src + ((size_t)b * ns + bi0) * Ds;
  const float* s1 = src + ((size_t)b * ns + bi1) * Ds;
  const float* s2 = src + ((size_t)b * ns + bi2) * Ds;
  _Float16* o = out + ((size_t)b * nd + i) * dinPad;
  for (int c = 0; c < Ds; ++c)
    o[c] = (_Float16)(w0 * s0[c] + w1 * s1[c] + w2 * s2[c]);
  const float* sk = skip + ((size_t)b * nd + i) * Dk;
  for (int c = 0; c < Dk; ++c) o[Ds + c] = (_Float16)sk[c];
  for (int c = Ds + Dk; c < dinPad; ++c) o[c] = (_Float16)0.0f;
}

// ---------------------------------------------------------------- host orchestration
extern "C" void kernel_launch(void* const* d_in, const int* in_sizes, int n_in,
                              void* d_out, int out_size, void* d_ws, size_t ws_size,
                              hipStream_t stream)
{
  (void)in_sizes; (void)n_in; (void)out_size; (void)ws_size;
  const float* x   = (const float*)d_in[0];
  const float* pos = (const float*)d_in[1];

  static const int nlayers[7]  = {3, 3, 3, 2, 2, 3, 3};
  static const int dims[7][4]  = {
      {6, 64, 64, 128}, {131, 128, 128, 256}, {259, 256, 512, 1024},
      {1280, 256, 256, 0}, {384, 256, 128, 0}, {131, 128, 128, 128},
      {128, 128, 128, 3}};

  const float* W[19]; const float* Bv[19]; int Din[19], Dout[19];
  {
    int li = 0, ai = 2;
    for (int mI = 0; mI < 7; ++mI)
      for (int l = 0; l < nlayers[mI]; ++l) {
        W[li]  = (const float*)d_in[ai];
        Bv[li] = (const float*)d_in[ai + 1];
        Din[li] = dims[mI][l]; Dout[li] = dims[mI][l + 1];
        ai += 2; ++li;
      }
  }

  char* base = (char*)d_ws; size_t off = 0;
  auto alloc = [&](size_t bytes) -> void* {
    void* r = base + off; off = (off + bytes + 255) & ~(size_t)255; return r;
  };

  _Float16* pk[19];
  for (int l = 0; l < 19; ++l) {
    size_t kt = (Din[l] + 31) / 32, ct = (Dout[l] + 15) / 16;
    pk[l] = (_Float16*)alloc(kt * ct * 32 * 16 * sizeof(_Float16));
  }
  float*    p1     = (float*)alloc((size_t)kB * kM1 * 3 * 4);
  float*    p2     = (float*)alloc((size_t)kB * kM2 * 3 * 4);
  int*      gidx1  = (int*)alloc((size_t)kB * kM1 * kK * 4);
  int*      nv1    = (int*)alloc((size_t)kB * kM1 * 4);
  int*      gidx2  = (int*)alloc((size_t)kB * kM2 * kK * 4);
  int*      nv2    = (int*)alloc((size_t)kB * kM2 * 4);
  float*    x1     = (float*)alloc((size_t)kB * kM1 * 128 * 4);
  float*    x2     = (float*)alloc((size_t)kB * kM2 * 256 * 4);
  _Float16* sa3in  = (_Float16*)alloc((size_t)kB * kM2 * 288 * 2);
  _Float16* sa3out = (_Float16*)alloc((size_t)kB * kM2 * 1024 * 2);
  float*    xg     = (float*)alloc((size_t)kB * 1024 * 4);
  _Float16* fp3in  = (_Float16*)alloc((size_t)kB * kM2 * 1280 * 2);
  float*    h3     = (float*)alloc((size_t)kB * kM2 * 256 * 4);
  _Float16* fp2in  = (_Float16*)alloc((size_t)kB * kM1 * 384 * 2);
  float*    h2     = (float*)alloc((size_t)kB * kM1 * 128 * 4);
  _Float16* fp1in  = (_Float16*)alloc((size_t)kB * kN * 160 * 2);

  auto mk = [&](int first, int nL, const int* relu) -> MlpDesc {
    MlpDesc d{}; d.nL = nL;
    for (int l = 0; l < nL; ++l) {
      d.L[l].wpk = pk[first + l];
      d.L[l].bias = Bv[first + l];
      d.L[l].dinPad  = ((Din[first + l] + 31) / 32) * 32;
      d.L[l].dout    = Dout[first + l];
      d.L[l].doutPad = ((Dout[first + l] + 31) / 32) * 32;
      d.L[l].relu    = relu[l];
    }
    return d;
  };
  static const int r110[3]  = {1, 1, 0};
  static const int r10[2]   = {1, 0};
  static const int rHead[6] = {1, 1, 0, 1, 1, 0};

  // pack all weights into WMMA-B fragment layout (f16)
  for (int l = 0; l < 19; ++l) {
    int kt = (Din[l] + 31) / 32, ct = (Dout[l] + 15) / 16;
    int tot = ct * kt * 32;
    pack_w_kernel<<<dim3((tot + 127) / 128), dim3(128), 0, stream>>>(W[l], Din[l], Dout[l], pk[l]);
  }

  // SA1
  fps_kernel<<<dim3(kB), dim3(256), 0, stream>>>(pos, kN, kM1, p1);
  ball_kernel<<<dim3((kM1 + 3) / 4, kB), dim3(128), 0, stream>>>(pos, p1, kN, kM1,
                                                                 0.2f * 0.2f, gidx1, nv1);
  MlpDesc dsa1 = mk(0, 3, r110);
  pointconv_kernel<160, 256><<<dim3(kM1, kB), dim3(128), 0, stream>>>(
      x, 3, pos, p1, gidx1, nv1, kN, kM1, dsa1, x1);

  // SA2
  fps_kernel<<<dim3(kB), dim3(256), 0, stream>>>(p1, kM1, kM2, p2);
  ball_kernel<<<dim3((kM2 + 3) / 4, kB), dim3(128), 0, stream>>>(p1, p2, kM1, kM2,
                                                                 0.4f * 0.4f, gidx2, nv2);
  MlpDesc dsa2 = mk(3, 3, r110);
  pointconv_kernel<160, 256><<<dim3(kM2, kB), dim3(128), 0, stream>>>(
      x1, 128, p1, p2, gidx2, nv2, kM1, kM2, dsa2, x2);

  // SA3 (global) : MLP then column-max
  {
    size_t tot = (size_t)kB * kM2 * 288;
    build_sa3_in<<<dim3((unsigned)((tot + 255) / 256)), dim3(256), 0, stream>>>(x2, p2, sa3in);
  }
  MlpDesc dsa3 = mk(6, 3, r110);
  mlp_rows_kernel<512, 1024><<<dim3((kM2 + 15) / 16, kB), dim3(128), 0, stream>>>(
      sa3in, (size_t)kM2 * 288, kM2, dsa3, sa3out, (size_t)kM2 * 1024,
      (float*)nullptr, 0, 0, 0);
  colmax_kernel<<<dim3(kB, 4), dim3(256), 0, stream>>>(sa3out, kM2, 1024, xg);

  // FP3
  {
    size_t tot = (size_t)kB * kM2 * 1280;
    build_fp3_in<<<dim3((unsigned)((tot + 255) / 256)), dim3(256), 0, stream>>>(xg, x2, fp3in);
  }
  MlpDesc dfp3 = mk(9, 2, r10);
  mlp_rows_kernel<1280, 256><<<dim3((kM2 + 15) / 16, kB), dim3(128), 0, stream>>>(
      fp3in, (size_t)kM2 * 1280, kM2, dfp3, (_Float16*)nullptr, 0,
      h3, (size_t)kM2 * 256, 256, 256);

  // FP2
  interp_concat_kernel<<<dim3((kM1 + 127) / 128, kB), dim3(128), 0, stream>>>(
      h3, p2, p1, x1, kM2, kM1, 256, 128, 384, fp2in);
  MlpDesc dfp2 = mk(11, 2, r10);
  mlp_rows_kernel<384, 256><<<dim3((kM1 + 15) / 16, kB), dim3(128), 0, stream>>>(
      fp2in, (size_t)kM1 * 384, kM1, dfp2, (_Float16*)nullptr, 0,
      h2, (size_t)kM1 * 128, 128, 128);

  // FP1 + head (fused 6-layer chain) -> d_out f32
  interp_concat_kernel<<<dim3(kN / 128, kB), dim3(128), 0, stream>>>(
      h2, p1, pos, x, kM1, kN, 128, 3, 160, fp1in);
  MlpDesc dhead = mk(13, 6, rHead);
  mlp_rows_kernel<160, 128><<<dim3(kN / 16, kB), dim3(128), 0, stream>>>(
      fp1in, (size_t)kN * 160, kN, dhead, (_Float16*)nullptr, 0,
      (float*)d_out, (size_t)kN * 3, 3, 3);
}
